// TinyRWKVStateMerger_85263690760276
// MI455X (gfx1250) — matrix operand order repaired and checked by reading, not compiled, not run on hardware
//
#include <hip/hip_runtime.h>

#define DD 32
#define HW 4096            // H*W
#define NROWS 128          // B*HP

typedef __attribute__((ext_vector_type(16))) _Float16 v16h;
typedef __attribute__((ext_vector_type(8)))  float    v8f;

__device__ __forceinline__ float sigmoidf_(float v) { return 1.0f / (1.0f + __expf(-v)); }

// ---------------- kernel 1: per-(b,l,hp) stats over H*W, then 12->32 projection ----------------
__global__ void stats_proj_kernel(const float* __restrict__ x,
                                  const float* __restrict__ in_proj_w,
                                  const float* __restrict__ in_proj_b,
                                  const float* __restrict__ layer_emb,
                                  float* __restrict__ featG) {
  const int p   = blockIdx.x;        // 0..4095 = b*1024 + (l*32+hp)
  const int b_  = p >> 10;
  const int rem = p & 1023;          // l*HP + hp
  const int l   = rem >> 5;
  const int hp  = rem & 31;
  const float* A  = x + ((size_t)b_ * 2048 + rem) * HW;
  const float* Bp = A + (size_t)1024 * HW;
  const int tid = threadIdx.x;

  float sa = 0.f, qa = 0.f, sb = 0.f, qb = 0.f, sab = 0.f;
  float mna = 3.4e38f, mxa = -3.4e38f, mnb = 3.4e38f, mxb = -3.4e38f;
#pragma unroll
  for (int it = 0; it < 4; ++it) {   // 4096 = 4 * (256 threads * 4 floats), b128 accesses
    const int base = it * 1024 + tid * 4;
    const float4 a4 = *(const float4*)(A + base);
    const float4 b4 = *(const float4*)(Bp + base);
    const float av[4] = {a4.x, a4.y, a4.z, a4.w};
    const float bv[4] = {b4.x, b4.y, b4.z, b4.w};
#pragma unroll
    for (int q = 0; q < 4; ++q) {
      sa += av[q]; qa += av[q] * av[q];
      sb += bv[q]; qb += bv[q] * bv[q];
      sab += av[q] * bv[q];
      mna = fminf(mna, av[q]); mxa = fmaxf(mxa, av[q]);
      mnb = fminf(mnb, bv[q]); mxb = fmaxf(mxb, bv[q]);
    }
  }
#pragma unroll
  for (int off = 16; off > 0; off >>= 1) {
    sa += __shfl_xor(sa, off); qa += __shfl_xor(qa, off);
    sb += __shfl_xor(sb, off); qb += __shfl_xor(qb, off);
    sab += __shfl_xor(sab, off);
    mna = fminf(mna, __shfl_xor(mna, off)); mxa = fmaxf(mxa, __shfl_xor(mxa, off));
    mnb = fminf(mnb, __shfl_xor(mnb, off)); mxb = fmaxf(mxb, __shfl_xor(mxb, off));
  }
  __shared__ float red[8][9];
  __shared__ float stats[12];
  const int wv = tid >> 5, ln = tid & 31;
  if (ln == 0) {
    red[wv][0] = sa;  red[wv][1] = qa;  red[wv][2] = sb;  red[wv][3] = qb; red[wv][4] = sab;
    red[wv][5] = mna; red[wv][6] = mxa; red[wv][7] = mnb; red[wv][8] = mxb;
  }
  __syncthreads();
  if (tid == 0) {
    float Sa = 0, Qa = 0, Sb = 0, Qb = 0, Sab = 0;
    float Mna = 3.4e38f, Mxa = -3.4e38f, Mnb = 3.4e38f, Mxb = -3.4e38f;
    for (int w = 0; w < 8; ++w) {
      Sa += red[w][0]; Qa += red[w][1]; Sb += red[w][2]; Qb += red[w][3]; Sab += red[w][4];
      Mna = fminf(Mna, red[w][5]); Mxa = fmaxf(Mxa, red[w][6]);
      Mnb = fminf(Mnb, red[w][7]); Mxb = fmaxf(Mxb, red[w][8]);
    }
    const float N = (float)HW, Nm1 = N - 1.0f;
    float sumd = Sa - Sb;
    float sqd  = fmaxf(Qa - 2.f * Sab + Qb, 0.f);
    stats[0]  = Sa / N;
    stats[1]  = sqrtf(fmaxf(Qa - Sa * Sa / N, 0.f) / Nm1);
    stats[2]  = Mna; stats[3] = Mxa;
    stats[4]  = Sb / N;
    stats[5]  = sqrtf(fmaxf(Qb - Sb * Sb / N, 0.f) / Nm1);
    stats[6]  = Mnb; stats[7] = Mxb;
    stats[8]  = sumd / N;
    stats[9]  = sqrtf(fmaxf(sqd - sumd * sumd / N, 0.f) / Nm1);
    stats[10] = sqrtf(sqd);
    stats[11] = Sab / (fmaxf(sqrtf(Qa), 1e-8f) * fmaxf(sqrtf(Qb), 1e-8f));
  }
  __syncthreads();
  if (tid < 32) {
    float acc = in_proj_b[tid] + layer_emb[l * DD + tid];
#pragma unroll
    for (int s = 0; s < 12; ++s) acc += stats[s] * in_proj_w[tid * 12 + s];
    // featG layout: [L, B*HP, D]
    featG[(size_t)l * (NROWS * DD) + (b_ * 32 + hp) * DD + tid] = acc;
  }
}

// ---------------- kernel 2: RWKV scan with WMMA for all small matmuls ----------------
// A-fragment (16-bit A 16x32 per ISA layout): lane<16 -> M=lane, K base {0..7,16..23}; lane>=16 -> +8 halves
__device__ __forceinline__ v16h load_a_frag(const _Float16* X, int w, int kc, int lane) {
  const int hf = lane >> 4, m = lane & 15;
  v16h a;
#pragma unroll
  for (int j = 0; j < 8; ++j) {
    int kA = ((j < 4) ? (2 * j) : (2 * j + 8)) + 8 * hf + kc * 32;
    a[2 * j]     = X[m * w + kA];
    a[2 * j + 1] = X[m * w + kA + 1];
  }
  return a;
}

__device__ __forceinline__ v16h load_b_frag(const _Float16* blob, int f, int lane) {
  return *(const v16h*)(blob + ((f * 32 + lane) << 4));
}

__global__ void __launch_bounds__(64)
rwkv_scan_kernel(float* __restrict__ featG,
                 const float* __restrict__ time_decay, const float* __restrict__ time_first,
                 const float* __restrict__ ln_tm_g, const float* __restrict__ ln_tm_b,
                 const float* __restrict__ tm_r, const float* __restrict__ tm_k,
                 const float* __restrict__ tm_v, const float* __restrict__ tm_out,
                 const float* __restrict__ ln_cm_g, const float* __restrict__ ln_cm_b,
                 const float* __restrict__ cm_k, const float* __restrict__ cm_v,
                 const float* __restrict__ cm_r) {
  __shared__ __align__(32) _Float16 wfrag[18 * 32 * 16];  // pre-packed B fragments
  __shared__ _Float16 sH[16 * 64];                        // xn / wkv / xn2 / kk f16 staging
  __shared__ float sF[512];                               // f, then y
  __shared__ float sR[512], sK[512], sV[512];
  __shared__ float sAa[512], sBb[512], sPp[512];
  __shared__ float sTf[32], sWd[32], sLtG[32], sLtB[32], sLcG[32], sLcB[32];

  const int tid  = threadIdx.x;
  const int wv   = tid >> 5;       // wave id -> N-tile
  const int lane = tid & 31;
  const int hf   = lane >> 4, nl = lane & 15;
  const int r0   = blockIdx.x * 16;  // 16 independent recurrence rows per block

  if (tid < 32) {
    sTf[tid] = time_first[tid];
    sWd[tid] = __expf(time_decay[tid]);
    sLtG[tid] = ln_tm_g[tid]; sLtB[tid] = ln_tm_b[tid];
    sLcG[tid] = ln_cm_g[tid]; sLcB[tid] = ln_cm_b[tid];
  }
  for (int i = tid; i < 512; i += 64) { sAa[i] = 0.f; sBb[i] = 0.f; sPp[i] = -1e30f; }

  // Pack weights once: B[k][n] = W[n][k] (xn @ W^T). Fragment ids:
  // 0-1 tm_r | 2-3 tm_k | 4-5 tm_v | 6-7 tm_out | 8-9 cm_r | 10-13 cm_k(N=64) | 14-17 cm_v(K=64,N=32)
  for (int idx = tid; idx < 18 * 512; idx += 64) {
    int f = idx >> 9, rem = idx & 511;
    int pl = rem >> 4, e = rem & 15;
    int j = e >> 1, t = e & 1;
    int ph = pl >> 4, pn = pl & 15;
    int k = (e & 7) + 8 * ph + 16 * (e >> 3);     // B-matrix K index, table order matching A
    float val;
    if (f < 8) {
      const float* W = (f < 2) ? tm_r : (f < 4) ? tm_k : (f < 6) ? tm_v : tm_out;
      val = W[((f & 1) * 16 + pn) * 32 + k];
    } else if (f < 10) {
      val = cm_r[((f - 8) * 16 + pn) * 32 + k];
    } else if (f < 14) {
      val = cm_k[((f - 10) * 16 + pn) * 32 + k];
    } else {
      int g = f - 14;
      val = cm_v[(((g >> 1) * 16) + pn) * 64 + (g & 1) * 32 + k];
    }
    wfrag[idx] = (_Float16)val;
  }
  __syncthreads();

  for (int l = 0; l < 32; ++l) {
    float* fg = featG + (size_t)l * (NROWS * DD) + r0 * DD;
    // prefetch next step's feature tile while this step computes (global_prefetch_b8)
    if (l < 31 && tid < 16) __builtin_prefetch(fg + NROWS * DD + tid * 32, 0, 3);
    for (int i = tid; i < 512; i += 64) sF[i] = fg[i];
    __syncthreads();
    if (tid < 16) {  // LN1 -> sH (f16)
      float mu = 0.f;
      for (int c = 0; c < 32; ++c) mu += sF[tid * 32 + c];
      mu *= (1.f / 32.f);
      float var = 0.f;
      for (int c = 0; c < 32; ++c) { float d = sF[tid * 32 + c] - mu; var += d * d; }
      float rs = rsqrtf(var * (1.f / 32.f) + 1e-5f);
      for (int c = 0; c < 32; ++c)
        sH[tid * 32 + c] = (_Float16)((sF[tid * 32 + c] - mu) * rs * sLtG[c] + sLtB[c]);
    }
    __syncthreads();
    {  // r,k,v = xn @ {tm_r,tm_k,tm_v}^T
      v16h a = load_a_frag(sH, 32, 0, lane);
      v8f z = {};
      v8f accR = __builtin_amdgcn_wmma_f32_16x16x32_f16(false, a, false, load_b_frag(wfrag, 0 + wv, lane), (short)0, z, false, false);
      v8f accK = __builtin_amdgcn_wmma_f32_16x16x32_f16(false, a, false, load_b_frag(wfrag, 2 + wv, lane), (short)0, z, false, false);
      v8f accV = __builtin_amdgcn_wmma_f32_16x16x32_f16(false, a, false, load_b_frag(wfrag, 4 + wv, lane), (short)0, z, false, false);
#pragma unroll
      for (int rr = 0; rr < 8; ++rr) {
        int o = (rr + 8 * hf) * 32 + wv * 16 + nl;
        sR[o] = accR[rr]; sK[o] = accK[rr]; sV[o] = accV[rr];
      }
    }
    __syncthreads();
    for (int i = tid; i < 512; i += 64) {  // elementwise WKV recurrence
      int c = i & 31;
      float kk = sK[i], vv = sV[i];
      float aa = sAa[i], bb = sBb[i], pp = sPp[i];
      float ww = sTf[c] + kk;
      float pmx = fmaxf(pp, ww);
      float e1 = __expf(pp - pmx), e2 = __expf(ww - pmx);
      float wkv = (e1 * aa + e2 * vv) / fmaxf(e1 * bb + e2, 1e-8f);
      sH[i] = (_Float16)wkv;
      float ww2 = pp - sWd[c];
      float p2 = fmaxf(ww2, kk);
      float e1b = __expf(ww2 - p2), e2b = __expf(kk - p2);
      sAa[i] = e1b * aa + e2b * vv;
      sBb[i] = e1b * bb + e2b;
      sPp[i] = p2;
    }
    __syncthreads();
    {  // y = f + sigmoid(r) * (wkv @ tm_out^T)
      v16h a = load_a_frag(sH, 32, 0, lane);
      v8f z = {};
      v8f accO = __builtin_amdgcn_wmma_f32_16x16x32_f16(false, a, false, load_b_frag(wfrag, 6 + wv, lane), (short)0, z, false, false);
#pragma unroll
      for (int rr = 0; rr < 8; ++rr) {
        int o = (rr + 8 * hf) * 32 + wv * 16 + nl;
        sF[o] += sigmoidf_(sR[o]) * accO[rr];
      }
    }
    __syncthreads();
    if (tid < 16) {  // LN2 -> sH
      float mu = 0.f;
      for (int c = 0; c < 32; ++c) mu += sF[tid * 32 + c];
      mu *= (1.f / 32.f);
      float var = 0.f;
      for (int c = 0; c < 32; ++c) { float d = sF[tid * 32 + c] - mu; var += d * d; }
      float rs = rsqrtf(var * (1.f / 32.f) + 1e-5f);
      for (int c = 0; c < 32; ++c)
        sH[tid * 32 + c] = (_Float16)((sF[tid * 32 + c] - mu) * rs * sLcG[c] + sLcB[c]);
    }
    __syncthreads();
    {  // channel-mix: kk = relu(xn2@cm_k^T)^2; y += sigmoid(xn2@cm_r^T) * (kk@cm_v^T)
      v16h a = load_a_frag(sH, 32, 0, lane);
      v8f z = {};
      v8f accK1 = __builtin_amdgcn_wmma_f32_16x16x32_f16(false, a, false, load_b_frag(wfrag, 10 + 2 * wv, lane), (short)0, z, false, false);
      v8f accK2 = __builtin_amdgcn_wmma_f32_16x16x32_f16(false, a, false, load_b_frag(wfrag, 11 + 2 * wv, lane), (short)0, z, false, false);
      v8f accCR = __builtin_amdgcn_wmma_f32_16x16x32_f16(false, a, false, load_b_frag(wfrag, 8 + wv, lane), (short)0, z, false, false);
      __syncthreads();  // everyone done reading xn2 from sH
#pragma unroll
      for (int rr = 0; rr < 8; ++rr) {
        int m = rr + 8 * hf;
        float q1 = fmaxf(accK1[rr], 0.f); q1 *= q1;
        float q2 = fmaxf(accK2[rr], 0.f); q2 *= q2;
        sH[m * 64 + wv * 32 + nl]      = (_Float16)q1;
        sH[m * 64 + wv * 32 + 16 + nl] = (_Float16)q2;
      }
      __syncthreads();  // kk ready (width-64 f16)
      v16h a0 = load_a_frag(sH, 64, 0, lane);
      v8f accV2 = __builtin_amdgcn_wmma_f32_16x16x32_f16(false, a0, false, load_b_frag(wfrag, 14 + 2 * wv, lane), (short)0, z, false, false);
      v16h a1 = load_a_frag(sH, 64, 1, lane);
      accV2 = __builtin_amdgcn_wmma_f32_16x16x32_f16(false, a1, false, load_b_frag(wfrag, 15 + 2 * wv, lane), (short)0, accV2, false, false);
#pragma unroll
      for (int rr = 0; rr < 8; ++rr) {
        int o = (rr + 8 * hf) * 32 + wv * 16 + nl;
        fg[o] = sF[o] + sigmoidf_(accCR[rr]) * accV2[rr];  // overwrite featG[l] with y
      }
    }
    __syncthreads();
  }
}

// ---------------- kernel 3: gate/row/col heads + sigmoid mask merge ----------------
__global__ void merge_kernel(const float* __restrict__ x, const float* __restrict__ feats,
                             const float* __restrict__ gate_w, const float* __restrict__ gate_b,
                             const float* __restrict__ row_w, const float* __restrict__ row_b,
                             const float* __restrict__ col_w, const float* __restrict__ col_b,
                             float* __restrict__ out, float* __restrict__ maskSums,
                             float* __restrict__ gateVals) {
  const int p = blockIdx.x;        // l*128 + row
  const int l = p >> 7, row = p & 127;
  const int b_ = row >> 5, hp = row & 31;
  const int pos = l * 32 + hp;
  const float* A  = x + ((size_t)b_ * 2048 + pos) * HW;
  const float* Bp = A + (size_t)1024 * HW;
  float* O = out + ((size_t)b_ * 1024 + pos) * HW;
  __shared__ float ft[32], rv[64], cv[64], gsh;
  const int tid = threadIdx.x;
  if (tid < 32) ft[tid] = feats[(size_t)l * (NROWS * DD) + row * DD + tid];
  __syncthreads();
  if (tid < 64) {
    float s = row_b[tid];
    for (int c = 0; c < 32; ++c) s += ft[c] * row_w[tid * 32 + c];
    rv[tid] = s;
  } else if (tid < 128) {
    int t = tid - 64;
    float s = col_b[t];
    for (int c = 0; c < 32; ++c) s += ft[c] * col_w[t * 32 + c];
    cv[t] = s;
  } else if (tid == 128) {
    float s = gate_b[0];
    for (int c = 0; c < 32; ++c) s += ft[c] * gate_w[c];
    gsh = s;
    gateVals[p] = sigmoidf_(s);
  }
  __syncthreads();
  const float g = gsh;
  float msum = 0.f;
#pragma unroll
  for (int it = 0; it < 4; ++it) {   // b128 path: 4 consecutive floats per thread
    const int i = it * 1024 + tid * 4;
    const float grh = g + rv[i >> 6];          // 4-aligned => same h for all 4 lanes' elems
    const float4 a4 = *(const float4*)(A + i);
    const float4 b4 = *(const float4*)(Bp + i);
    const int w0 = i & 63;
    float m0 = sigmoidf_(grh + cv[w0 + 0]);
    float m1 = sigmoidf_(grh + cv[w0 + 1]);
    float m2 = sigmoidf_(grh + cv[w0 + 2]);
    float m3 = sigmoidf_(grh + cv[w0 + 3]);
    float4 o4;
    o4.x = m0 * a4.x + (1.f - m0) * b4.x;
    o4.y = m1 * a4.y + (1.f - m1) * b4.y;
    o4.z = m2 * a4.z + (1.f - m2) * b4.z;
    o4.w = m3 * a4.w + (1.f - m3) * b4.w;
    *(float4*)(O + i) = o4;
    msum += (m0 + m1) + (m2 + m3);
  }
#pragma unroll
  for (int off = 16; off > 0; off >>= 1) msum += __shfl_xor(msum, off);
  __shared__ float mr[8];
  if ((tid & 31) == 0) mr[tid >> 5] = msum;
  __syncthreads();
  if (tid == 0) {
    float s = 0.f;
    for (int w = 0; w < 8; ++w) s += mr[w];
    maskSums[p] = s;
  }
}

// ---------------- kernel 4: deterministic final reduction of the two means ----------------
__global__ void finalize_kernel(const float* __restrict__ maskSums,
                                const float* __restrict__ gateVals,
                                float* __restrict__ out2) {
  const int tid = threadIdx.x;
  float ms = 0.f, gs = 0.f;
  for (int i = tid; i < 4096; i += 256) { ms += maskSums[i]; gs += gateVals[i]; }
#pragma unroll
  for (int off = 16; off > 0; off >>= 1) { ms += __shfl_xor(ms, off); gs += __shfl_xor(gs, off); }
  __shared__ float mr[8], gr[8];
  if ((tid & 31) == 0) { mr[tid >> 5] = ms; gr[tid >> 5] = gs; }
  __syncthreads();
  if (tid == 0) {
    float S = 0.f, G = 0.f;
    for (int w = 0; w < 8; ++w) { S += mr[w]; G += gr[w]; }
    out2[0] = G / 4096.f;
    out2[1] = S / (4096.f * 4096.f);
  }
}

extern "C" void kernel_launch(void* const* d_in, const int* in_sizes, int n_in,
                              void* d_out, int out_size, void* d_ws, size_t ws_size,
                              hipStream_t stream) {
  (void)in_sizes; (void)n_in; (void)ws_size;
  const float* x          = (const float*)d_in[0];
  const float* in_proj_w  = (const float*)d_in[1];
  const float* in_proj_b  = (const float*)d_in[2];
  const float* layer_emb  = (const float*)d_in[3];
  const float* time_decay = (const float*)d_in[4];
  const float* time_first = (const float*)d_in[5];
  const float* ln_tm_g    = (const float*)d_in[6];
  const float* ln_tm_b    = (const float*)d_in[7];
  const float* tm_r       = (const float*)d_in[8];
  const float* tm_k       = (const float*)d_in[9];
  const float* tm_v       = (const float*)d_in[10];
  const float* tm_out     = (const float*)d_in[11];
  const float* ln_cm_g    = (const float*)d_in[12];
  const float* ln_cm_b    = (const float*)d_in[13];
  const float* cm_k       = (const float*)d_in[14];
  const float* cm_v       = (const float*)d_in[15];
  const float* cm_r       = (const float*)d_in[16];
  const float* gate_w     = (const float*)d_in[17];
  const float* gate_b     = (const float*)d_in[18];
  const float* row_w      = (const float*)d_in[19];
  const float* row_b      = (const float*)d_in[20];
  const float* col_w      = (const float*)d_in[21];
  const float* col_b      = (const float*)d_in[22];

  float* ws       = (float*)d_ws;
  float* featG    = ws;                   // 32*128*32 = 131072 floats (scan overwrites in place with y)
  float* maskSums = ws + 131072;          // 4096 floats
  float* gateVals = ws + 131072 + 4096;   // 4096 floats
  float* outF     = (float*)d_out;

  stats_proj_kernel<<<4096, 256, 0, stream>>>(x, in_proj_w, in_proj_b, layer_emb, featG);
  rwkv_scan_kernel<<<8, 64, 0, stream>>>(featG, time_decay, time_first, ln_tm_g, ln_tm_b,
                                         tm_r, tm_k, tm_v, tm_out, ln_cm_g, ln_cm_b,
                                         cm_k, cm_v, cm_r);
  merge_kernel<<<4096, 256, 0, stream>>>(x, featG, gate_w, gate_b, row_w, row_b, col_w, col_b,
                                         outF, maskSums, gateVals);
  finalize_kernel<<<1, 256, 0, stream>>>(maskSums, gateVals, outF + (out_size - 2));
}
